// Pct_seg_10419590660688
// MI455X (gfx1250) — matrix operand order repaired and checked by reading, not compiled, not run on hardware
//
#include <hip/hip_runtime.h>

// ---------------------------------------------------------------------------
// PCT-seg forward for gfx1250 (MI455X): bf16 WMMA GEMMs with fused epilogues.
// Block tile 64(M) x 128(N), K-step 32, 8 wave32 waves, double-buffered LDS
// (one barrier per K-step), 4 v_wmma per wave per K-step with A reuse,
// incremental tile pointers, LDS row pitch 40 elems (80B) -> bank-conflict-free
// fragment loads on the 64-bank x 4B LDS.
// ---------------------------------------------------------------------------

typedef __bf16 bf16;
typedef __attribute__((ext_vector_type(8)))  bf16  v8bf;
typedef __attribute__((ext_vector_type(16))) bf16  v16bf;
typedef __attribute__((ext_vector_type(8)))  float v8f;

#define BN_INV_STD 0.9999950000374997f   // 1/sqrt(1 + 1e-5)
#define LDS_PITCH  40                    // 32 data + 8 pad bf16 (80B, 16B-aligned rows)

struct GemmP {
  const bf16* A;        // aT==0: [M][aStride] row-major; aT==1: [K][aStride] (A^T layout)
  long long   abs_;     // A batch stride (elements), 0 for shared weights
  int         aT;
  int         aStride;
  const bf16* X;        // [K][N] row-major per batch
  long long   xbs;      // X batch stride
  float*      outF;     // optional f32 out
  long long   obsF;     // f32 batch stride
  long long   soM, soN; // f32 store strides (lets s3 store transposed [N,50])
  bf16*       outB;     // optional bf16 out, row stride = N
  long long   obsB;
  int M, N, K, Mreal;
  const float* cb;      // conv bias per-M (nullable)
  const float* g;       // BN gamma per-M (nullable -> scale 1)
  const float* bb;      // BN beta per-M (nullable)
  const float* colscale;// per-N scale [B][N] (nullable) -- attention L1 renorm
  const float* res;     // residual add [B][M][N] (nullable)
  long long   rbs;
  int act;              // 0 none, 1 relu, 2 leaky(0.2)
};

__device__ __forceinline__ v16bf make16(v8bf lo, v8bf hv) {
  return __builtin_shufflevector(lo, hv, 0,1,2,3,4,5,6,7,8,9,10,11,12,13,14,15);
}

template <int AT>
__device__ __forceinline__ void gemm_body(const GemmP& p) {
  __shared__ bf16 lA[2][64][LDS_PITCH];
  __shared__ bf16 lB[2][128][LDS_PITCH];   // transposed: [n][k]

  const int tid  = threadIdx.x;
  const int wave = tid >> 5;
  const int lane = tid & 31;
  const int wm = wave >> 1, wn = wave & 1;  // wave tile: 16 rows x 64 cols
  const int lr = lane & 15;
  const int hi = lane >> 4;
  const int b  = blockIdx.z;
  const int n0 = blockIdx.x << 7;
  const int m0 = blockIdx.y << 6;

  const bf16* Ab = p.A + (long long)b * p.abs_;
  const bf16* Xb = p.X + (long long)b * p.xbs;

  // staging coordinates (256 threads cover each tile exactly once)
  const int sAr = tid >> 2, sAk = (tid & 3) << 3;  // AT==0: A[64][32], 8 elems/thread
  const int sTk = tid >> 3, sTm = (tid & 7) << 3;  // AT==1: transpose-stage
  const int sBk = tid >> 3, sBn = (tid & 7) << 4;  // B[32][128] -> lB[n][k], 16 elems/thread

  // incremental per-thread tile pointers (advance by constant stride per K-step)
  const bf16* pA  = (AT == 0)
      ? (Ab + (long long)(m0 + sAr) * p.aStride + sAk)
      : (Ab + (long long)sTk * p.aStride + m0 + sTm);
  const long long stepA = (AT == 0) ? 32LL : 32LL * p.aStride;
  const bf16* pB  = Xb + (long long)sBk * p.N + n0 + sBn;
  const long long stepB = 32LL * (long long)p.N;

  v8f acc[4] = {v8f{}, v8f{}, v8f{}, v8f{}};

  // ---- preload + stage tile 0 ----
  v8bf ra  = *(const v8bf*)(pA);
  v8bf rb0 = *(const v8bf*)(pB);
  v8bf rb1 = *(const v8bf*)(pB + 8);
  pA += stepA; pB += stepB;
  if (AT == 0) {
    *(v8bf*)(&lA[0][sAr][sAk]) = ra;
  } else {
#pragma unroll
    for (int e = 0; e < 8; ++e) lA[0][sTm + e][sTk] = ra[e];
  }
#pragma unroll
  for (int e = 0; e < 8; ++e) lB[0][sBn + e][sBk] = rb0[e];
#pragma unroll
  for (int e = 0; e < 8; ++e) lB[0][sBn + 8 + e][sBk] = rb1[e];
  __syncthreads();

  const int nsteps = p.K >> 5;
  for (int i = 0; i < nsteps; ++i) {
    const int cur = i & 1, nxt = cur ^ 1;
    const bool more = (i + 1) < nsteps;

    if (more) {  // next-tile global loads overlap the WMMAs below
      ra  = *(const v8bf*)(pA);
      rb0 = *(const v8bf*)(pB);
      rb1 = *(const v8bf*)(pB + 8);
    }
    // speculative prefetch one tile beyond next (bad translations dropped)
    __builtin_prefetch(pB + stepB, 0, 1);

    // ---- compute from current buffer ----
    // A 16x32 (16-bit): lanes0-15 row=lr K{0..7,16..23}; lanes16-31 K{8..15,24..31}
    const bf16* Ar = &lA[cur][(wm << 4) + lr][hi << 3];
    const v16bf af = make16(*(const v8bf*)(Ar), *(const v8bf*)(Ar + 16));
    // load all four B fragments, then the 4 WMMAs (A operand reused)
    v16bf bj[4];
#pragma unroll
    for (int j = 0; j < 4; ++j) {
      // B 32x16: lanes0-15 col=lr K{0..7,16..23}; lanes16-31 col=lr K{8..15,24..31}
      const bf16* Br = &lB[cur][(wn << 6) + (j << 4) + lr][hi << 3];
      bj[j] = make16(*(const v8bf*)(Br), *(const v8bf*)(Br + 16));
    }
#pragma unroll
    for (int j = 0; j < 4; ++j)
      acc[j] = __builtin_amdgcn_wmma_f32_16x16x32_bf16(false, af, false, bj[j],
                                                       (short)0, acc[j], false, false);

    if (more) {  // stage next buffer, then one barrier per K-step
      if (AT == 0) {
        *(v8bf*)(&lA[nxt][sAr][sAk]) = ra;
      } else {
#pragma unroll
        for (int e = 0; e < 8; ++e) lA[nxt][sTm + e][sTk] = ra[e];
      }
#pragma unroll
      for (int e = 0; e < 8; ++e) lB[nxt][sBn + e][sBk] = rb0[e];
#pragma unroll
      for (int e = 0; e < 8; ++e) lB[nxt][sBn + 8 + e][sBk] = rb1[e];
      pA += stepA; pB += stepB;
    }
    __syncthreads();
  }

  // ---- epilogue: C 16x16 f32: VGPR r -> M = r + 8*hi, N = lr ----
  const int ncol0 = n0 + (wn << 6) + lr;
  float cs[4] = {1.f, 1.f, 1.f, 1.f};
  if (p.colscale) {
#pragma unroll
    for (int j = 0; j < 4; ++j) cs[j] = p.colscale[(long long)b * p.N + ncol0 + (j << 4)];
  }
  const long long obF = (long long)b * p.obsF;
  const long long obB = (long long)b * p.obsB;
  const long long obR = (long long)b * p.rbs;
#pragma unroll
  for (int r = 0; r < 8; ++r) {
    const int m = m0 + (wm << 4) + (hi << 3) + r;
    if (m >= p.Mreal) continue;
    const float sc  = p.g  ? p.g[m] * BN_INV_STD : 1.f;
    const float cbv = p.cb ? p.cb[m] : 0.f;
    const float bbv = p.bb ? p.bb[m] : 0.f;
#pragma unroll
    for (int j = 0; j < 4; ++j) {
      const int n = ncol0 + (j << 4);
      float v = (acc[j][r] * cs[j] + cbv) * sc + bbv;
      if (p.act == 1)      v = fmaxf(v, 0.f);
      else if (p.act == 2) v = v > 0.f ? v : 0.2f * v;
      if (p.res)  v += p.res[obR + (long long)m * p.N + n];
      if (p.outF) p.outF[obF + (long long)m * p.soM + (long long)n * p.soN] = v;
      if (p.outB) p.outB[obB + (long long)m * p.N + n] = (bf16)v;
    }
  }
}

__launch_bounds__(256)
__global__ void gemm_wmma_bf16(GemmP p) {
  if (p.aT == 0) gemm_body<0>(p);
  else           gemm_body<1>(p);
}

// ---- helper kernels -------------------------------------------------------

// f32 [M,K] -> bf16 [Mp,Kp], zero-padded
__global__ void cvt_pad(const float* in, bf16* out, int M, int K, int Mp, int Kp) {
  long long i = (long long)blockIdx.x * 256 + threadIdx.x;
  if (i >= (long long)Mp * Kp) return;
  int m = (int)(i / Kp), k = (int)(i % Kp);
  out[i] = (m < M && k < K) ? (bf16)in[(long long)m * K + k] : (bf16)0.f;
}

// x [B,N,3] f32 -> xt [B,32,N] bf16 (channels 3..31 zero)
__global__ void prep_x(const float* x, bf16* xt, int N) {
  long long i = (long long)blockIdx.x * 256 + threadIdx.x;
  if (i >= (long long)2 * 32 * N) return;
  int n = (int)(i % N), c = (int)((i / N) % 32), b = (int)(i / ((long long)32 * N));
  xt[i] = (c < 3) ? (bf16)x[((long long)b * N + n) * 3 + c] : (bf16)0.f;
}

// row softmax: energy f32 [B,N,N] -> att bf16 [B,N,N]
__global__ void softmax_row(const float* E, bf16* A, int N) {
  const int n = blockIdx.x, b = blockIdx.z, tid = threadIdx.x;
  const float* row = E + ((long long)b * N + n) * (long long)N;
  bf16* arow      = A + ((long long)b * N + n) * (long long)N;
  __shared__ float red[256];
  float mx = -3.4e38f;
  for (int m = tid; m < N; m += 256) mx = fmaxf(mx, row[m]);
  red[tid] = mx; __syncthreads();
  for (int s = 128; s > 0; s >>= 1) { if (tid < s) red[tid] = fmaxf(red[tid], red[tid + s]); __syncthreads(); }
  mx = red[0]; __syncthreads();
  float sum = 0.f;
  for (int m = tid; m < N; m += 256) sum += __expf(row[m] - mx);
  red[tid] = sum; __syncthreads();
  for (int s = 128; s > 0; s >>= 1) { if (tid < s) red[tid] += red[tid + s]; __syncthreads(); }
  const float inv = 1.f / red[0];
  for (int m = tid; m < N; m += 256) arow[m] = (bf16)(__expf(row[m] - mx) * inv);
}

// colscale[b][m] = 1 / (1e-9 + sum_n att[b][n][m])
__global__ void colsum_scale(const bf16* A, float* cs, int N) {
  const int m = blockIdx.x * 256 + threadIdx.x, b = blockIdx.z;
  if (m >= N) return;
  float s = 0.f;
  const bf16* base = A + (long long)b * N * N + m;
  for (int n = 0; n < N; ++n) s += (float)base[(long long)n * N];
  cs[(long long)b * N + m] = 1.f / (1e-9f + s);
}

// o = bf16(x - xr), elementwise
__global__ void sub_bf16(const float* x, const float* xr, bf16* o, long long cnt) {
  long long i = (long long)blockIdx.x * 256 + threadIdx.x;
  if (i < cnt) o[i] = (bf16)(x[i] - xr[i]);
}

// per-(b,c) max over n of face, broadcast into scat[b][C+c][:]
__global__ void maxpool_bcast(const float* face, bf16* scat, int C, int N) {
  const int c = blockIdx.x, b = blockIdx.z, tid = threadIdx.x;
  const float* row = face + ((long long)b * C + c) * (long long)N;
  __shared__ float red[256];
  float mx = -3.4e38f;
  for (int n = tid; n < N; n += 256) mx = fmaxf(mx, row[n]);
  red[tid] = mx; __syncthreads();
  for (int s = 128; s > 0; s >>= 1) { if (tid < s) red[tid] = fmaxf(red[tid], red[tid + s]); __syncthreads(); }
  const bf16 v = (bf16)red[0];
  bf16* out = scat + ((long long)b * (2 * C) + C + c) * (long long)N;
  for (int n = tid; n < N; n += 256) out[n] = v;
}

// ---- host orchestration ---------------------------------------------------

extern "C" void kernel_launch(void* const* d_in, const int* in_sizes, int n_in,
                              void* d_out, int out_size, void* d_ws, size_t ws_size,
                              hipStream_t stream) {
  (void)in_sizes; (void)n_in; (void)out_size; (void)ws_size;
  const int N = 4096; const long long LN = N;

  const float* IN[40];
  for (int i = 0; i < 40; ++i) IN[i] = (const float*)d_in[i];
  const float* x = IN[0];

  size_t off = 0;
  auto alloc = [&](size_t bytes) -> void* {
    void* p = (char*)d_ws + off;
    off += (bytes + 255) & ~(size_t)255;
    return p;
  };
  auto cvt = [&](const float* w, int M, int K, int Mp, int Kp) -> bf16* {
    bf16* o = (bf16*)alloc((size_t)Mp * Kp * sizeof(bf16));
    long long tot = (long long)Mp * Kp;
    cvt_pad<<<dim3((unsigned)((tot + 255) / 256)), dim3(256), 0, stream>>>(w, o, M, K, Mp, Kp);
    return o;
  };
  auto gemm = [&](GemmP p) {
    gemm_wmma_bf16<<<dim3(p.N / 128, p.M / 64, 2), dim3(256), 0, stream>>>(p);
  };
  auto G = [&](const bf16* A, int aStride, const bf16* X, long long xbs,
               int M, int K) -> GemmP {
    GemmP p{};
    p.A = A; p.aStride = aStride; p.aT = 0; p.abs_ = 0;
    p.X = X; p.xbs = xbs;
    p.M = M; p.N = N; p.K = K; p.Mreal = M;
    p.soM = N; p.soN = 1;
    return p;
  };

  // weights -> padded bf16
  bf16* w1   = cvt(IN[1], 64, 3, 64, 32);
  bf16* w2   = cvt(IN[4], 128, 64, 128, 64);
  bf16* w3   = cvt(IN[7], 256, 128, 256, 128);
  bf16* wp1  = cvt(IN[10], 256, 256, 256, 256);
  bf16* wp2  = cvt(IN[13], 256, 256, 256, 256);
  bf16* wqk[2] = { cvt(IN[16], 128, 256, 128, 256), cvt(IN[23], 128, 256, 128, 256) };
  bf16* wv[2]  = { cvt(IN[17], 256, 256, 256, 256), cvt(IN[24], 256, 256, 256, 256) };
  bf16* wt[2]  = { cvt(IN[19], 256, 256, 256, 256), cvt(IN[26], 256, 256, 256, 256) };
  bf16* wcf  = cvt(IN[30], 512, 1024, 512, 1024);
  bf16* ws1  = cvt(IN[33], 512, 1024, 512, 1024);
  bf16* ws2  = cvt(IN[36], 256, 512, 256, 512);
  bf16* ws3  = cvt(IN[39], 50, 256, 64, 256);

  // activations / attention scratch
  bf16*  xt    = (bf16*)alloc(2 * 32 * LN * 2);
  bf16*  hb1   = (bf16*)alloc(2 * 64 * LN * 2);
  bf16*  hb2   = (bf16*)alloc(2 * 128 * LN * 2);
  bf16*  hb3   = (bf16*)alloc(2 * 256 * LN * 2);
  bf16*  hp1   = (bf16*)alloc(2 * 256 * LN * 2);
  float* hp2f  = (float*)alloc(2 * 256 * LN * 4);
  bf16*  xqb   = (bf16*)alloc(2 * 128 * LN * 2);
  float* energ = (float*)alloc(2 * LN * LN * 4);
  bf16*  attB  = (bf16*)alloc(2 * LN * LN * 2);
  float* colsc = (float*)alloc(2 * LN * 4);
  bf16*  xvb   = (bf16*)alloc(2 * 256 * LN * 2);
  float* xrF   = (float*)alloc(2 * 256 * LN * 4);
  bf16*  tinB  = (bf16*)alloc(2 * 256 * LN * 2);
  bf16*  xcat  = (bf16*)alloc(2 * 1024 * LN * 2);
  float* sfA   = (float*)alloc(2 * 256 * LN * 4);
  float* sfB   = (float*)alloc(2 * 256 * LN * 4);
  float* faceF = (float*)alloc(2 * 512 * LN * 4);
  bf16*  scatB = (bf16*)alloc(2 * 1024 * LN * 2);
  bf16*  s1B   = (bf16*)alloc(2 * 512 * LN * 2);
  bf16*  s2B   = (bf16*)alloc(2 * 256 * LN * 2);

  // input transpose + pad: [B,N,3] -> [B,32,N] bf16
  prep_x<<<dim3((unsigned)((2 * 32 * LN + 255) / 256)), dim3(256), 0, stream>>>(x, xt, N);

  // conv1..conv3, pt1, pt2
  { GemmP p = G(w1, 32, xt, 32 * LN, 64, 32);
    p.g = IN[2]; p.bb = IN[3]; p.act = 1; p.outB = hb1; p.obsB = 64 * LN; gemm(p); }
  { GemmP p = G(w2, 64, hb1, 64 * LN, 128, 64);
    p.g = IN[5]; p.bb = IN[6]; p.act = 1; p.outB = hb2; p.obsB = 128 * LN; gemm(p); }
  { GemmP p = G(w3, 128, hb2, 128 * LN, 256, 128);
    p.g = IN[8]; p.bb = IN[9]; p.act = 1; p.outB = hb3; p.obsB = 256 * LN; gemm(p); }
  { GemmP p = G(wp1, 256, hb3, 256 * LN, 256, 256);
    p.g = IN[11]; p.bb = IN[12]; p.act = 1; p.outB = hp1; p.obsB = 256 * LN; gemm(p); }
  { GemmP p = G(wp2, 256, hp1, 256 * LN, 256, 256);
    p.g = IN[14]; p.bb = IN[15]; p.act = 1;
    p.outF = hp2f; p.obsF = 256 * LN; p.outB = hb3 /*reuse as bf16 of h*/; p.obsB = 256 * LN; gemm(p); }

  // 4 offset-attention layers (layer 0: sa1 params; layers 1..3: sa2 params)
  const float* curF = hp2f;
  const bf16*  curX = hb3; long long curXbs = 256 * LN;
  float* ping[2] = { sfA, sfB };
  for (int l = 0; l < 4; ++l) {
    const int pi = (l == 0) ? 0 : 1;
    const float* vb = (pi == 0) ? IN[18] : IN[25];
    const float* tb = (pi == 0) ? IN[20] : IN[27];
    const float* sg = (pi == 0) ? IN[21] : IN[28];
    const float* sb = (pi == 0) ? IN[22] : IN[29];

    // xq = qk_w @ x  -> bf16 [B,128,N]
    { GemmP p = G(wqk[pi], 256, curX, curXbs, 128, 256);
      p.outB = xqb; p.obsB = 128 * LN; gemm(p); }
    // energy = xq^T xq  -> f32 [B,N,N]   (A transposed-layout path)
    { GemmP p = G(xqb, N, xqb, 128 * LN, N, 128);
      p.aT = 1; p.abs_ = 128 * LN;
      p.outF = energ; p.obsF = LN * LN; gemm(p); }
    softmax_row<<<dim3(N, 1, 2), dim3(256), 0, stream>>>(energ, attB, N);
    colsum_scale<<<dim3(N / 256, 1, 2), dim3(256), 0, stream>>>(attB, colsc, N);
    // xv = v_w @ x + v_b  -> bf16
    { GemmP p = G(wv[pi], 256, curX, curXbs, 256, 256);
      p.cb = vb; p.outB = xvb; p.obsB = 256 * LN; gemm(p); }
    // x_r = (xv @ att) * colscale  -> f32
    { GemmP p = G(xvb, N, attB, LN * LN, 256, N);
      p.abs_ = 256 * LN; p.colscale = colsc;
      p.outF = xrF; p.obsF = 256 * LN; gemm(p); }
    // t_in = bf16(x - x_r)
    sub_bf16<<<dim3((unsigned)((2 * 256 * LN + 255) / 256)), dim3(256), 0, stream>>>(
        curF, xrF, tinB, 2 * 256 * LN);
    // out = x + relu(bn(t_w @ t_in + t_b)); bf16 copy goes into xcat slice l
    float* nxtF = ping[l & 1];
    { GemmP p = G(wt[pi], 256, tinB, 256 * LN, 256, 256);
      p.cb = tb; p.g = sg; p.bb = sb; p.act = 1;
      p.res = curF; p.rbs = 256 * LN;
      p.outF = nxtF; p.obsF = 256 * LN;
      p.outB = xcat + (long long)l * 256 * LN; p.obsB = 1024 * LN; gemm(p); }
    curF = nxtF;
    curX = xcat + (long long)l * 256 * LN; curXbs = 1024 * LN;
  }

  // conv_fuse: face = leaky(bn(cf_w @ xcat)); bf16 into lower half of scat
  { GemmP p = G(wcf, 1024, xcat, 1024 * LN, 512, 1024);
    p.g = IN[31]; p.bb = IN[32]; p.act = 2;
    p.outF = faceF; p.obsF = 512 * LN;
    p.outB = scatB; p.obsB = 1024 * LN; gemm(p); }
  // global max-pool broadcast into upper half of scat
  maxpool_bcast<<<dim3(512, 1, 2), dim3(256), 0, stream>>>(faceF, scatB, 512, N);

  { GemmP p = G(ws1, 1024, scatB, 1024 * LN, 512, 1024);
    p.g = IN[34]; p.bb = IN[35]; p.act = 2; p.outB = s1B; p.obsB = 512 * LN; gemm(p); }
  { GemmP p = G(ws2, 512, s1B, 512 * LN, 256, 512);
    p.g = IN[37]; p.bb = IN[38]; p.act = 2; p.outB = s2B; p.obsB = 256 * LN; gemm(p); }
  // s3: [50,256] @ s2 -> d_out [B,N,50] (transposed f32 store)
  { GemmP p = G(ws3, 256, s2B, 256 * LN, 64, 256);
    p.Mreal = 50;
    p.outF = (float*)d_out; p.obsF = LN * 50; p.soM = 1; p.soN = 50; gemm(p); }
}
